// PartialSequenceDiffusion_3685081940028
// MI455X (gfx1250) — matrix operand order repaired and checked by reading, not compiled, not run on hardware
//
#include <hip/hip_runtime.h>
#include <hip/hip_bf16.h>

typedef _Float16 h16;
typedef __attribute__((ext_vector_type(16))) _Float16 v16h;
typedef __attribute__((ext_vector_type(8)))  _Float16 v8h;
typedef __attribute__((ext_vector_type(8)))  float    v8f;

#define NN 8192
#define DD 256
#define KK 32
#define PP 64
#define HH 8

// ---------------- helpers ----------------

__device__ __forceinline__ float wave_sum(float v) {
#pragma unroll
  for (int o = 16; o > 0; o >>= 1) v += __shfl_xor(v, o, 32);
  return v;
}
__device__ __forceinline__ float wave_max(float v) {
#pragma unroll
  for (int o = 16; o > 0; o >>= 1) v = fmaxf(v, __shfl_xor(v, o, 32));
  return v;
}
__device__ __forceinline__ float gelu_f(float x) {
  float x3 = x * x * x;
  return 0.5f * x * (1.0f + tanhf(0.7978845608028654f * (x + 0.044715f * x3)));
}

// A fragment: 16x32 f16, row-major source (lda halfs).
// lanes 0-15: row = lane,  K = k0+0..7  and k0+16..23
// lanes16-31: row = lane-16,K = k0+8..15 and k0+24..31
__device__ __forceinline__ v16h load_a_frag(const h16* A, int lda, int row0, int k0, int lane) {
  int r  = row0 + (lane & 15);
  int kb = k0 + ((lane >> 4) << 3);
  const h16* p = A + (size_t)r * lda + kb;
  v8h lo = *(const v8h*)p;
  v8h hi = *(const v8h*)(p + 16);
  v16h a;
#pragma unroll
  for (int i = 0; i < 8; ++i) { a[i] = lo[i]; a[i + 8] = hi[i]; }
  return a;
}

// B fragment from pre-packed buffer: frag = tn*ksteps+ks, 512 halfs per frag.
__device__ __forceinline__ v16h load_b_frag(const h16* Bp, int frag, int lane) {
  const h16* p = Bp + (size_t)frag * 512 + lane * 16;
  v8h lo = *(const v8h*)p;
  v8h hi = *(const v8h*)(p + 8);
  v16h b;
#pragma unroll
  for (int i = 0; i < 8; ++i) { b[i] = lo[i]; b[i + 8] = hi[i]; }
  return b;
}

__device__ __forceinline__ v8f wmma_step(v8f c, v16h a, v16h b) {
  return __builtin_amdgcn_wmma_f32_16x16x32_f16(false, a, false, b, (short)0, c, false, false);
}

// ---------------- kernels ----------------

// pack W[K,Mreal] f32 row-major -> B-fragment layout halves, padded to Mpad cols
__global__ void k_pack(const float* __restrict__ W, h16* __restrict__ dst,
                       int Kdim, int Mreal, int Mpad) {
  int t = blockIdx.x * blockDim.x + threadIdx.x;
  int lane = t & 31, rest = t >> 5;
  int ksteps = Kdim >> 5;
  int ks = rest % ksteps, tn = rest / ksteps;
  if (tn >= (Mpad >> 4)) return;
  int col = tn * 16 + (lane & 15);
  int kbase = ks * 32 + ((lane >> 4) << 3);
  h16* d = dst + ((size_t)(tn * ksteps + ks) * 32 + lane) * 16;
#pragma unroll
  for (int j = 0; j < 16; ++j) {
    float v = (col < Mreal) ? W[(size_t)(kbase + j + ((j >> 3) << 3)) * Mreal + col] : 0.0f;
    d[j] = (h16)v;
  }
}

// two LayerNorms of features -> f16 (one block per node, 256 threads)
__global__ void k_ln(const float* __restrict__ x,
                     const float* __restrict__ g1, const float* __restrict__ b1,
                     const float* __restrict__ g2, const float* __restrict__ b2,
                     h16* __restrict__ f1h, h16* __restrict__ f2h) {
  int n = blockIdx.x, t = threadIdx.x;
  float v = x[(size_t)n * DD + t];
  __shared__ float s1[8], s2[8];
  __shared__ float mu_s, rs_s;
  float ws = wave_sum(v), wq = wave_sum(v * v);
  if ((t & 31) == 0) { s1[t >> 5] = ws; s2[t >> 5] = wq; }
  __syncthreads();
  if (t == 0) {
    float a = 0.f, b = 0.f;
#pragma unroll
    for (int i = 0; i < 8; ++i) { a += s1[i]; b += s2[i]; }
    float mu = a / DD, var = b / DD - mu * mu;
    mu_s = mu; rs_s = rsqrtf(var + 1e-5f);
  }
  __syncthreads();
  float xc = (v - mu_s) * rs_s;
  f1h[(size_t)n * DD + t] = (h16)(g1[t] * xc + b1[t]);
  f2h[(size_t)n * DD + t] = (h16)(g2[t] * xc + b2[t]);
}

// generic WMMA GEMM: C[nrows,mcols] = act(A[nrows,kdim] @ Bp + bias) + residual
// One wave computes a 32x64 tile: 2 A-frags x 4 B-frags -> 8 independent WMMAs
// per K-step (hides XDL hazard windows, 4x less A traffic than 16x16 tiling).
__global__ void k_gemm(const h16* __restrict__ A, const h16* __restrict__ Bp,
                       const float* __restrict__ bias, const float* __restrict__ residual,
                       float* __restrict__ C, h16* __restrict__ Ch,
                       int nrows, int mcols, int kdim, int do_gelu) {
  int wid = (blockIdx.x * blockDim.x + threadIdx.x) >> 5;
  int lane = threadIdx.x & 31;
  int tilesN4 = mcols >> 6;                 // groups of 4 col-tiles (64 cols)
  int total = (nrows >> 5) * tilesN4;       // 32-row tiles
  if (wid >= total) return;
  int tm = wid / tilesN4, tn4 = wid % tilesN4;
  int ksteps = kdim >> 5;
  v8f acc[8];
#pragma unroll
  for (int c = 0; c < 8; ++c)
#pragma unroll
    for (int i = 0; i < 8; ++i) acc[c][i] = 0.0f;

  for (int ks = 0; ks < ksteps; ++ks) {
    v16h a0 = load_a_frag(A, kdim, tm * 32,      ks * 32, lane);
    v16h a1 = load_a_frag(A, kdim, tm * 32 + 16, ks * 32, lane);
#pragma unroll
    for (int c = 0; c < 4; ++c) {
      v16h b = load_b_frag(Bp, (tn4 * 4 + c) * ksteps + ks, lane);
      acc[c]     = wmma_step(acc[c],     a0, b);
      acc[4 + c] = wmma_step(acc[4 + c], a1, b);
    }
  }

#pragma unroll
  for (int c = 0; c < 4; ++c) {
    int col = (tn4 * 4 + c) * 16 + (lane & 15);
    float bi = bias ? bias[col] : 0.0f;
#pragma unroll
    for (int half = 0; half < 2; ++half) {
      int rbase = tm * 32 + half * 16 + ((lane >> 4) << 3);
      v8f av = acc[half * 4 + c];
#pragma unroll
      for (int r = 0; r < 8; ++r) {
        float v = av[r] + bi;
        if (do_gelu) v = gelu_f(v);
        size_t idx = (size_t)(rbase + r) * mcols + col;
        if (residual) v += residual[idx];
        if (C)  C[idx]  = v;
        if (Ch) Ch[idx] = (h16)v;
      }
    }
  }
}

// fused per-edge pair features + LN + MLP(64->128->64) + pair-bias(64->8)
// one block = 64 edge rows, 8 waves
__global__ void k_pair(const float* __restrict__ wrel, const float* __restrict__ left,
                       const float* __restrict__ right,
                       const float* __restrict__ g_lnp, const float* __restrict__ b_lnp,
                       const float* __restrict__ b_m1, const float* __restrict__ b_m2,
                       const h16* __restrict__ wm1p, const h16* __restrict__ wm2p,
                       const h16* __restrict__ wpbp,
                       const int* __restrict__ neighbours, const int* __restrict__ resi,
                       const int* __restrict__ chain, const int* __restrict__ batch,
                       float* __restrict__ pairbias) {
  __shared__ float p0[64][64];
  __shared__ __align__(16) h16 p0h[64][64];
  __shared__ __align__(16) h16 hbuf[64][128];
  int t = threadIdx.x;
  int r0 = blockIdx.x * 64;

  { // stage 1: build raw pair features
    int row = t >> 2, seg = t & 3;
    int r = r0 + row, n = r >> 5, kk = r & 31;
    int nbr = neighbours[n * KK + kk];
    int nbv = nbr < 0 ? 0 : nbr;
    int rel = resi[n] - resi[nbv];
    rel = rel < -32 ? -32 : (rel > 32 ? 32 : rel);
    rel += 32;
    bool same = (chain[n] == chain[nbv]) && (batch[n] == batch[nbv]);
#pragma unroll
    for (int c = 0; c < 16; ++c) {
      int col = seg * 16 + c;
      float v = left[(size_t)n * PP + col] + right[(size_t)nbv * PP + col];
      if (same) v += wrel[rel * PP + col];
      p0[row][col] = v;
    }
  }
  __syncthreads();
  if (t < 64) { // stage 2: LayerNorm per row -> f16
    float s = 0.f, s2 = 0.f;
#pragma unroll
    for (int c = 0; c < 64; ++c) { float x = p0[t][c]; s += x; s2 += x * x; }
    float mu = s / 64.f, var = s2 / 64.f - mu * mu;
    float rs = rsqrtf(var + 1e-5f);
#pragma unroll
    for (int c = 0; c < 64; ++c)
      p0h[t][c] = (h16)(g_lnp[c] * ((p0[t][c] - mu) * rs) + b_lnp[c]);
  }
  __syncthreads();
  int wave = t >> 5, lane = t & 31;
  { // stage 3: MLP1 64->128, GELU
    int rt = wave >> 1, ch = wave & 1;
#pragma unroll
    for (int ct = 0; ct < 4; ++ct) {
      int tn = ch * 4 + ct;
      v8f acc;
#pragma unroll
      for (int i = 0; i < 8; ++i) acc[i] = 0.0f;
#pragma unroll
      for (int ks = 0; ks < 2; ++ks) {
        v16h a = load_a_frag(&p0h[0][0], 64, rt * 16, ks * 32, lane);
        v16h b = load_b_frag(wm1p, tn * 2 + ks, lane);
        acc = wmma_step(acc, a, b);
      }
      int col = tn * 16 + (lane & 15);
#pragma unroll
      for (int rr = 0; rr < 8; ++rr) {
        float v = gelu_f(acc[rr] + b_m1[col]);
        hbuf[rt * 16 + rr + ((lane >> 4) << 3)][col] = (h16)v;
      }
    }
  }
  __syncthreads();
  { // stage 4: MLP2 128->64 (write back into p0h as f16)
    int rt = wave >> 1, ch = wave & 1;
#pragma unroll
    for (int c2 = 0; c2 < 2; ++c2) {
      int tn = ch * 2 + c2;
      v8f acc;
#pragma unroll
      for (int i = 0; i < 8; ++i) acc[i] = 0.0f;
#pragma unroll
      for (int ks = 0; ks < 4; ++ks) {
        v16h a = load_a_frag(&hbuf[0][0], 128, rt * 16, ks * 32, lane);
        v16h b = load_b_frag(wm2p, tn * 4 + ks, lane);
        acc = wmma_step(acc, a, b);
      }
      int col = tn * 16 + (lane & 15);
#pragma unroll
      for (int rr = 0; rr < 8; ++rr)
        p0h[rt * 16 + rr + ((lane >> 4) << 3)][col] = (h16)(acc[rr] + b_m2[col]);
    }
  }
  __syncthreads();
  if (wave < 4) { // stage 5: pair bias 64->8 (padded to 16)
    int rt = wave;
    v8f acc;
#pragma unroll
    for (int i = 0; i < 8; ++i) acc[i] = 0.0f;
#pragma unroll
    for (int ks = 0; ks < 2; ++ks) {
      v16h a = load_a_frag(&p0h[0][0], 64, rt * 16, ks * 32, lane);
      v16h b = load_b_frag(wpbp, ks, lane);
      acc = wmma_step(acc, a, b);
    }
    int col = lane & 15;
    if (col < 8) {
#pragma unroll
      for (int rr = 0; rr < 8; ++rr) {
        int rg = r0 + rt * 16 + rr + ((lane >> 4) << 3);
        int n = rg >> 5, kk = rg & 31;
        pairbias[((size_t)n * HH + col) * KK + kk] = acc[rr];
      }
    }
  }
}

// sparse neighbour attention; one block per node, one wave per head
__global__ void k_attn(const float* __restrict__ qf, const float* __restrict__ kf,
                       const float* __restrict__ vf, const float* __restrict__ pairbias,
                       const float* __restrict__ mask, const int* __restrict__ neighbours,
                       h16* __restrict__ attnh) {
  int n = blockIdx.x, t = threadIdx.x;
  int h = t >> 5, d = t & 31;
  __shared__ float qs[DD];
  __shared__ int nbs[KK];
  __shared__ float pms[KK];
  qs[t] = qf[(size_t)n * DD + t];
  if (t < KK) {
    int nbr = neighbours[n * KK + t];
    int nbv = nbr < 0 ? 0 : nbr;
    nbs[t] = nbv;
    pms[t] = mask[n] * mask[nbv] * (nbr != -1 ? 1.0f : 0.0f);
  }
  __syncthreads();
  float qd = qs[h * 32 + d];
  float mylogit = 0.0f;
  for (int k = 0; k < KK; ++k) {
    float kvv = kf[(size_t)nbs[k] * DD + h * 32 + d];
    float s = wave_sum(qd * kvv);
    if (d == k)
      mylogit = s * 0.17677669529663687f + pairbias[((size_t)n * HH + h) * KK + k];
  }
  float pm = pms[d];
  float lm = pm > 0.0f ? mylogit : -1e9f;
  float m = wave_max(lm);
  float e = pm > 0.0f ? __expf(lm - m) : 0.0f;
  float s = wave_sum(e);
  float att = e / fmaxf(s, 1e-20f);
  float acc = 0.0f;
  for (int k = 0; k < KK; ++k) {
    float ak = __shfl(att, k, 32);
    acc += ak * vf[(size_t)nbs[k] * DD + h * 32 + d];
  }
  attnh[(size_t)n * DD + t] = (h16)acc;
}

// batch segment mean of bias_pre[N,512] -> meanb[4,512]
__global__ void k_segmean(const float* __restrict__ bias_pre, const int* __restrict__ batch,
                          const float* __restrict__ mask, float* __restrict__ meanb) {
  int b = blockIdx.x, t = threadIdx.x;
  float a0 = 0.f, a1 = 0.f, cnt = 0.f;
  for (int n = 0; n < NN; ++n) {
    if (batch[n] != b) continue;
    float m = mask[n];
    a0 += bias_pre[(size_t)n * 512 + t] * m;
    a1 += bias_pre[(size_t)n * 512 + 256 + t] * m;
    cnt += m;
  }
  float c = fmaxf(cnt, 1.0f);
  meanb[b * 512 + t] = a0 / c;
  meanb[b * 512 + 256 + t] = a1 / c;
}

// per-(segment, head) masked mean of keyg (x) valg outer products
__global__ void k_segop(const float* __restrict__ keygf, const float* __restrict__ valgf,
                        const int* __restrict__ chain, const int* __restrict__ batch,
                        const float* __restrict__ mask,
                        float* __restrict__ chain_op, float* __restrict__ batch_op) {
  int sb = blockIdx.x;
  int seg = sb >> 3, h = sb & 7;
  const int* arr; float* out; int segid;
  if (seg < 16) { arr = chain; out = chain_op + (size_t)(seg * HH + h) * 1024; segid = seg; }
  else          { arr = batch; out = batch_op + (size_t)((seg - 16) * HH + h) * 1024; segid = seg - 16; }
  __shared__ float acc[32][32];
  __shared__ float kb[32], vb[32];
  __shared__ float cnt;
  int t = threadIdx.x;
  float* af = &acc[0][0];
  af[t] = 0.f; af[t + 256] = 0.f; af[t + 512] = 0.f; af[t + 768] = 0.f;
  if (t == 0) cnt = 0.f;
  __syncthreads();
  for (int n = 0; n < NN; ++n) {
    if (arr[n] != segid) continue;       // block-uniform branch
    float m = mask[n];
    if (t < 32)      kb[t]      = keygf[(size_t)n * DD + h * 32 + t] * m;
    else if (t < 64) vb[t - 32] = valgf[(size_t)n * DD + h * 32 + (t - 32)];
    if (t == 0) cnt += m;
    __syncthreads();
    int i = t >> 3, j0 = (t & 7) * 4;
    float ki = kb[i];
#pragma unroll
    for (int jj = 0; jj < 4; ++jj) acc[i][j0 + jj] += ki * vb[j0 + jj];
    __syncthreads();
  }
  float c = fmaxf(cnt, 1.0f);
  out[t] = af[t] / c; out[t + 256] = af[t + 256] / c;
  out[t + 512] = af[t + 512] / c; out[t + 768] = af[t + 768] / c;
}

// gout[n, h*64 + v] = op . qg + biasg  (f16 rows for final GEMM)
__global__ void k_gout(const float* __restrict__ qgf, const float* __restrict__ chain_op,
                       const float* __restrict__ batch_op, const float* __restrict__ meanb,
                       const int* __restrict__ chain, const int* __restrict__ batch,
                       h16* __restrict__ gouth) {
  int n = blockIdx.x, t = threadIdx.x;
  int h = t >> 5, d = t & 31;
  __shared__ float qs[DD];
  qs[t] = qgf[(size_t)n * DD + t];
  __syncthreads();
  int c = chain[n], b = batch[n];
  const float* cop = chain_op + (size_t)(c * HH + h) * 1024 + d * 32;
  const float* bop = batch_op + (size_t)(b * HH + h) * 1024 + d * 32;
  float a1 = 0.f, a2 = 0.f;
#pragma unroll
  for (int k = 0; k < 32; ++k) {
    float q = qs[h * 32 + k];
    a1 += cop[k] * q;
    a2 += bop[k] * q;
  }
  int col1 = h * 64 + d, col2 = col1 + 32;
  gouth[(size_t)n * 512 + col1] = (h16)(a1 + meanb[b * 512 + col1]);
  gouth[(size_t)n * 512 + col2] = (h16)(a2 + meanb[b * 512 + col2]);
}

// ---------------- host ----------------

extern "C" void kernel_launch(void* const* d_in, const int* in_sizes, int n_in,
                              void* d_out, int out_size, void* d_ws, size_t ws_size,
                              hipStream_t stream) {
  const float* features = (const float*)d_in[0];
  const float* mask     = (const float*)d_in[1];
  const float* w_relpos = (const float*)d_in[2];
  const float* w_left   = (const float*)d_in[3];
  const float* w_right  = (const float*)d_in[4];
  const float* g_ln1    = (const float*)d_in[5];
  const float* b_ln1    = (const float*)d_in[6];
  const float* g_lnp    = (const float*)d_in[7];
  const float* b_lnp    = (const float*)d_in[8];
  const float* w_m1     = (const float*)d_in[9];
  const float* b_m1     = (const float*)d_in[10];
  const float* w_m2     = (const float*)d_in[11];
  const float* b_m2     = (const float*)d_in[12];
  const float* g_ln2    = (const float*)d_in[13];
  const float* b_ln2    = (const float*)d_in[14];
  const float* wq       = (const float*)d_in[15];
  const float* wk       = (const float*)d_in[16];
  const float* wv       = (const float*)d_in[17];
  const float* w_pb     = (const float*)d_in[18];
  const float* wo       = (const float*)d_in[19];
  const float* wkg      = (const float*)d_in[20];
  const float* bkg      = (const float*)d_in[21];
  const float* wvg      = (const float*)d_in[22];
  const float* bvg      = (const float*)d_in[23];
  const float* wqg      = (const float*)d_in[24];
  const float* bqg      = (const float*)d_in[25];
  const float* wbg      = (const float*)d_in[26];
  const float* bbg      = (const float*)d_in[27];
  const float* wog      = (const float*)d_in[28];
  const int* neighbours = (const int*)d_in[29];
  const int* resi       = (const int*)d_in[30];
  const int* chain      = (const int*)d_in[31];
  const int* batch      = (const int*)d_in[32];
  float* out = (float*)d_out;

  // ---- workspace bump allocator (phase-aliased) ----
  char* ws = (char*)d_ws;
  size_t off = 0;
  auto alloc = [&](size_t bytes) -> char* {
    char* p = ws + off;
    off = (off + bytes + 255) & ~(size_t)255;
    return p;
  };
  // phase1 qkv -> phase2 keyg/valg/qg (same slots)
  float* q_f = (float*)alloc((size_t)NN * DD * 4);
  float* k_f = (float*)alloc((size_t)NN * DD * 4);
  float* v_f = (float*)alloc((size_t)NN * DD * 4);
  float* keyg_f = q_f; float* valg_f = k_f; float* qg_f = v_f;
  // bias_pre (16MB), pairbias aliases its first 8MB
  float* bias_pre = (float*)alloc((size_t)NN * 512 * 4);
  float* pairbias = bias_pre;
  // gouth (8MB), f1h/f2h alias it
  h16* gouth = (h16*)alloc((size_t)NN * 512 * 2);
  h16* f1h = gouth;
  h16* f2h = gouth + (size_t)NN * DD;
  // attnh
  h16* attnh = (h16*)alloc((size_t)NN * DD * 2);
  // left/right (phase1) aliased with fmh (phase2)
  char* lr = alloc((size_t)NN * DD * 2); // 4MB
  float* left  = (float*)lr;
  float* right = (float*)(lr + (size_t)NN * PP * 4);
  h16* fmh = (h16*)lr;
  // small stats
  float* meanb    = (float*)alloc(4 * 512 * 4);
  float* chain_op = (float*)alloc((size_t)16 * HH * 1024 * 4);
  float* batch_op = (float*)alloc((size_t)4 * HH * 1024 * 4);
  // packed weights
  h16* wl_p  = (h16*)alloc(256 * 64 * 2);
  h16* wr_p  = (h16*)alloc(256 * 64 * 2);
  h16* wq_p  = (h16*)alloc(256 * 256 * 2);
  h16* wk_p  = (h16*)alloc(256 * 256 * 2);
  h16* wv_p  = (h16*)alloc(256 * 256 * 2);
  h16* wo_p  = (h16*)alloc(256 * 256 * 2);
  h16* wkg_p = (h16*)alloc(256 * 256 * 2);
  h16* wvg_p = (h16*)alloc(256 * 256 * 2);
  h16* wqg_p = (h16*)alloc(256 * 256 * 2);
  h16* wm1_p = (h16*)alloc(64 * 128 * 2);
  h16* wm2_p = (h16*)alloc(128 * 64 * 2);
  h16* wpb_p = (h16*)alloc(64 * 16 * 2);
  h16* wbg_p = (h16*)alloc(256 * 512 * 2);
  h16* wog_p = (h16*)alloc(512 * 256 * 2);

  auto pack = [&](const float* W, h16* dst, int Kd, int Mr, int Mp) {
    int threads = (Mp / 16) * (Kd / 32) * 32;
    int blocks = (threads + 255) / 256;
    k_pack<<<blocks, 256, 0, stream>>>(W, dst, Kd, Mr, Mp);
  };
  auto gemm = [&](const h16* A, const h16* Bp, const float* bias, const float* res,
                  float* C, h16* Ch, int nrows, int mcols, int kdim, int gl) {
    int tiles = (nrows / 32) * (mcols / 64);   // 32x64 per wave
    int blocks = (tiles + 7) / 8;
    k_gemm<<<blocks, 256, 0, stream>>>(A, Bp, bias, res, C, Ch, nrows, mcols, kdim, gl);
  };

  // 1) pack all weights into WMMA B-fragment layout
  pack(w_left, wl_p, 256, 64, 64);   pack(w_right, wr_p, 256, 64, 64);
  pack(wq, wq_p, 256, 256, 256);     pack(wk, wk_p, 256, 256, 256);
  pack(wv, wv_p, 256, 256, 256);     pack(wo, wo_p, 256, 256, 256);
  pack(wkg, wkg_p, 256, 256, 256);   pack(wvg, wvg_p, 256, 256, 256);
  pack(wqg, wqg_p, 256, 256, 256);
  pack(w_m1, wm1_p, 64, 128, 128);   pack(w_m2, wm2_p, 128, 64, 64);
  pack(w_pb, wpb_p, 64, 8, 16);
  pack(wbg, wbg_p, 256, 512, 512);   pack(wog, wog_p, 512, 256, 256);

  // 2) LayerNorms -> f16 activations
  k_ln<<<NN, 256, 0, stream>>>(features, g_ln1, b_ln1, g_ln2, b_ln2, f1h, f2h);

  // 3) pair-input projections and q/k/v
  gemm(f1h, wl_p, nullptr, nullptr, left, nullptr, NN, 64, 256, 0);
  gemm(f1h, wr_p, nullptr, nullptr, right, nullptr, NN, 64, 256, 0);
  gemm(f2h, wq_p, nullptr, nullptr, q_f, nullptr, NN, 256, 256, 0);
  gemm(f2h, wk_p, nullptr, nullptr, k_f, nullptr, NN, 256, 256, 0);
  gemm(f2h, wv_p, nullptr, nullptr, v_f, nullptr, NN, 256, 256, 0);

  // 4) fused per-edge pair MLP -> pair bias logits
  k_pair<<<(NN * KK) / 64, 256, 0, stream>>>(w_relpos, left, right, g_lnp, b_lnp,
                                             b_m1, b_m2, wm1_p, wm2_p, wpb_p,
                                             neighbours, resi, chain, batch, pairbias);

  // 5) sparse attention -> f16 attention output
  k_attn<<<NN, 256, 0, stream>>>(q_f, k_f, v_f, pairbias, mask, neighbours, attnh);

  // 6) features_mid = features + attn @ wo  (f32 to d_out, f16 to fmh)
  gemm(attnh, wo_p, nullptr, features, out, fmh, NN, 256, 256, 0);

  // 7) global projections
  gemm(fmh, wkg_p, bkg, nullptr, keyg_f, nullptr, NN, 256, 256, 1);
  gemm(fmh, wvg_p, bvg, nullptr, valg_f, nullptr, NN, 256, 256, 0);
  gemm(fmh, wqg_p, bqg, nullptr, qg_f, nullptr, NN, 256, 256, 1);
  gemm(fmh, wbg_p, bbg, nullptr, bias_pre, nullptr, NN, 512, 256, 0);

  // 8) segment statistics
  k_segmean<<<4, 256, 0, stream>>>(bias_pre, batch, mask, meanb);
  k_segop<<<(16 + 4) * 8, 256, 0, stream>>>(keyg_f, valg_f, chain, batch, mask,
                                            chain_op, batch_op);

  // 9) per-node global attention readout -> f16 gout rows
  k_gout<<<NN, 256, 0, stream>>>(qg_f, chain_op, batch_op, meanb, chain, batch, gouth);

  // 10) final: d_out += gout @ wog
  gemm(gouth, wog_p, nullptr, out, out, nullptr, NN, 256, 512, 0);
}
